// HierarchicalLiftedStructureLoss_42082089566281
// MI455X (gfx1250) — compile-verified, matrix operand order checked by reading, not run on hardware
//
#include <hip/hip_runtime.h>
#include <hip/hip_bf16.h>
#include <math.h>

#define DEV __device__ __forceinline__

typedef __attribute__((ext_vector_type(16))) __bf16         v16bf;
typedef __attribute__((ext_vector_type(8)))  float          v8f;
typedef __attribute__((ext_vector_type(8)))  unsigned short v8u16;
typedef __attribute__((ext_vector_type(16))) unsigned short v16u16;

constexpr int   BATCH = 8;
constexpr int   N     = 2048;
constexpr int   D     = 128;
constexpr float NEGF  = -1e30f;

// ---------- small helpers ----------
DEV unsigned short f32_to_bf16(float f) {
  unsigned int u = __float_as_uint(f);
  unsigned int r = u + 0x7fffu + ((u >> 16) & 1u);   // round-to-nearest-even
  return (unsigned short)(r >> 16);
}
DEV float bf16_to_f32(unsigned short h) {
  return __uint_as_float(((unsigned int)h) << 16);
}

DEV float wave_sum_f(float v) {
  for (int o = 16; o; o >>= 1) v += __shfl_xor(v, o);
  return v;
}
DEV int wave_sum_i(int v) {
  for (int o = 16; o; o >>= 1) v += __shfl_xor(v, o);
  return v;
}
// max with smallest-index tie break (matches jnp.argmax first-occurrence)
DEV void wave_argmax(float& d, int& idx) {
  for (int o = 16; o; o >>= 1) {
    float od = __shfl_xor(d, o);
    int   oi = __shfl_xor(idx, o);
    if (od > d || (od == d && oi < idx)) { d = od; idx = oi; }
  }
}

// sorted-ascending keep-K-smallest insertion (fully unrolled -> stays in VGPRs)
template <int K>
DEV void insert_sorted(float (&a)[K], float v) {
  if (v < a[K - 1]) {
    a[K - 1] = v;
#pragma unroll
    for (int i = K - 1; i > 0; --i) {
      float lo = fminf(a[i - 1], a[i]);
      float hi = fmaxf(a[i - 1], a[i]);
      a[i - 1] = lo; a[i] = hi;
    }
  }
}

// merge 32 per-lane sorted lists into the global K smallest (all lanes get result)
template <int K>
DEV void extract_topk(float (&a)[K], int lane, float (&out)[K]) {
#pragma unroll
  for (int i = 0; i < K; ++i) {
    float v = a[0];
    int   l = lane;
    for (int o = 16; o; o >>= 1) {
      float ov = __shfl_xor(v, o);
      int   ol = __shfl_xor(l, o);
      if (ov < v || (ov == v && ol < l)) { v = ov; l = ol; }
    }
    out[i] = v;
    if (l == lane) {
#pragma unroll
      for (int j = 0; j < K - 1; ++j) a[j] = a[j + 1];
      a[K - 1] = INFINITY;
    }
  }
}

template <int K>
DEV float lse_k(const float (&kv)[K], int kvalid, float p) {
  float m = NEGF;
#pragma unroll
  for (int i = 0; i < K; ++i) {
    float av = (i < kvalid) ? fabsf(p - kv[i]) : NEGF;
    m = fmaxf(m, av);
  }
  float s = 0.0f;
#pragma unroll
  for (int i = 0; i < K; ++i)
    if (i < kvalid) s += __expf(fabsf(p - kv[i]) - m);
  return m + __logf(fmaxf(s, 1e-30f));
}

// Load one 16-bit WMMA operand fragment per the CDNA5 wave32 layout:
// lane L (half = L>>4, lr = L&15) of row (base row + lr) holds
// K = k0..k0+7 and K = k0+16..k0+23, where k0 = kchunk*32 + half*8.
DEV v16bf load_frag(const unsigned short* __restrict__ base, int row, int k0) {
  const unsigned short* p = base + row * D + k0;
  v8u16 a = *(const v8u16*)(p);
  v8u16 b = *(const v8u16*)(p + 16);
  v16u16 cat = __builtin_shufflevector(a, b, 0, 1, 2, 3, 4, 5, 6, 7,
                                       8, 9, 10, 11, 12, 13, 14, 15);
  return __builtin_bit_cast(v16bf, cat);
}

// ---------- K0a: split x into bf16 hi/lo (bf16x3 f32 emulation) ----------
__global__ void k_split(const float* __restrict__ x, unsigned short* __restrict__ xhi,
                        unsigned short* __restrict__ xlo, int total) {
  int i = blockIdx.x * blockDim.x + threadIdx.x;
  if (i >= total) return;
  float v = x[i];
  unsigned short h = f32_to_bf16(v);
  xhi[i] = h;
  xlo[i] = f32_to_bf16(v - bf16_to_f32(h));
}

// ---------- K0b: per-row squared norms ----------
__global__ void k_sqnorm(const float* __restrict__ x, float* __restrict__ sq) {
  int wave = (blockIdx.x * blockDim.x + threadIdx.x) >> 5;
  int lane = threadIdx.x & 31;
  if (wave >= BATCH * N) return;
  const float* row = x + (size_t)wave * D;
  float s = 0.f;
  for (int j = lane; j < D; j += 32) { float v = row[j]; s += v * v; }
  s = wave_sum_f(s);
  if (lane == 0) sq[wave] = s;
}

// ---------- K1: fused WMMA dist panel (LDS-resident) + row selection ----------
__global__ void __launch_bounds__(256) k_panel(
    const unsigned short* __restrict__ xhi, const unsigned short* __restrict__ xlo,
    const float* __restrict__ sq, const int* __restrict__ t,
    float* __restrict__ lse2, float* __restrict__ lse4,
    int* __restrict__ cnt2o, int* __restrict__ cnt4o,
    float* __restrict__ pospair, int* __restrict__ rro, int* __restrict__ hpo) {
  extern __shared__ float sdist[];                 // 16 x N f32 = 128 KB (<320KB/WGP)
  const int b       = blockIdx.x >> 7;             // N/16 = 128 row-blocks per batch
  const int rowBase = (blockIdx.x & 127) * 16;
  const int tid  = threadIdx.x;
  const int w    = tid >> 5, lane = tid & 31;
  const int half = lane >> 4, lr = lane & 15;

  const unsigned short* xhib = xhi + (size_t)b * N * D;
  const unsigned short* xlob = xlo + (size_t)b * N * D;

  // A fragments (this block's 16 rows, full K=128), kept live in VGPRs.
  v16bf ahi[4], alo[4];
#pragma unroll
  for (int kc = 0; kc < 4; ++kc) {
    int k0 = kc * 32 + half * 8;
    ahi[kc] = load_frag(xhib, rowBase + lr, k0);
    alo[kc] = load_frag(xlob, rowBase + lr, k0);
  }

  // Each wave sweeps 16 of the 128 column tiles. bf16x3: hi*hi + hi*lo + lo*hi.
  for (int it = 0; it < 16; ++it) {
    int colBase = (w + it * 8) * 16;
    v8f acc = {0.f, 0.f, 0.f, 0.f, 0.f, 0.f, 0.f, 0.f};
#pragma unroll
    for (int kc = 0; kc < 4; ++kc) {
      int k0 = kc * 32 + half * 8;
      v16bf bhi = load_frag(xhib, colBase + lr, k0);
      v16bf blo = load_frag(xlob, colBase + lr, k0);
      acc = __builtin_amdgcn_wmma_f32_16x16x32_bf16(false, ahi[kc], false, bhi,
                                                    (short)0, acc, false, false);
      acc = __builtin_amdgcn_wmma_f32_16x16x32_bf16(false, ahi[kc], false, blo,
                                                    (short)0, acc, false, false);
      acc = __builtin_amdgcn_wmma_f32_16x16x32_bf16(false, alo[kc], false, bhi,
                                                    (short)0, acc, false, false);
    }
    // C/D layout: vgpr r, lane half 0 -> M=r ; half 1 -> M=r+8 ; N = lr
#pragma unroll
    for (int r = 0; r < 8; ++r)
      sdist[(r + half * 8) * N + colBase + lr] = acc[r];
  }
  __syncthreads();

  // ----- selection: 8 waves x 2 rows, dist panel read from LDS -----
  const float* sqb = sq + (size_t)b * N;
  for (int sub = 0; sub < 2; ++sub) {
    const int m    = w + sub * 8;
    const int grow = rowBase + m;
    const float* arow = sdist + m * N;
    const int*   trow = t + ((size_t)b * N + grow) * N;
    const float  sqm  = sqb[grow];

    __builtin_prefetch(trow + lane * 64, 0, 1);    // cover the 8KB t row

    // pass 1: hardest positive (t==0 && dist>0), first-occurrence argmax
    float best = NEGF; int bestIdx = N;
    for (int j = lane; j < N; j += 32) {
      float dd = sqm + sqb[j] - 2.0f * arow[j];
      int   tv = trow[j];
      if (tv == 0 && dd > 0.0f && dd > best) { best = dd; bestIdx = j; }
    }
    wave_argmax(best, bestIdx);
    const int haspos = (best > 0.0f) ? 1 : 0;
    const int jstar  = haspos ? bestIdx : 0;

    // pass 2: rank (#positives before jstar) + per-penalty top-k smallest
    float top2[8], top4[7];
#pragma unroll
    for (int i = 0; i < 8; ++i) top2[i] = INFINITY;
#pragma unroll
    for (int i = 0; i < 7; ++i) top4[i] = INFINITY;
    int c2 = 0, c4 = 0, rk = 0;
    for (int j = lane; j < N; j += 32) {
      float dd = sqm + sqb[j] - 2.0f * arow[j];
      int   tv = trow[j];
      if (tv == 0 && dd > 0.0f && j < jstar) rk++;
      if (tv == 2)      { c2++; insert_sorted<8>(top2, dd); }
      else if (tv == 4) { c4++; insert_sorted<7>(top4, dd); }
    }
    rk = wave_sum_i(rk);
    c2 = wave_sum_i(c2);
    c4 = wave_sum_i(c4);

    float kv2[8], kv4[7];
    extract_topk<8>(top2, lane, kv2);
    extract_topk<7>(top4, lane, kv4);
    float l2 = (c2 > 0) ? lse_k<8>(kv2, (c2 < 8 ? c2 : 8), 2.0f) : 0.0f;
    float l4 = (c4 > 0) ? lse_k<7>(kv4, (c4 < 7 ? c4 : 7), 4.0f) : 0.0f;

    if (lane == 0) {
      size_t gi  = (size_t)b * N + grow;
      lse2[gi]    = l2;  lse4[gi]  = l4;
      cnt2o[gi]   = c2;  cnt4o[gi] = c4;
      pospair[gi] = haspos ? best : 0.0f;
      rro[gi]     = rk;                  // == clip(rank, 0, N-1)
      hpo[gi]     = haspos;
    }
  }
}

// ---------- K3: gather at rank r, per-batch masked mean ----------
__global__ void k_combine(const float* __restrict__ lse2, const float* __restrict__ lse4,
                          const int* __restrict__ cnt2, const int* __restrict__ cnt4,
                          const float* __restrict__ pospair, const int* __restrict__ rro,
                          const int* __restrict__ hpo, float* __restrict__ batchloss) {
  __shared__ float sJ[256];
  __shared__ int   sC[256];
  const int b = blockIdx.x, tid = threadIdx.x;
  float accJ = 0.f; int accC = 0;
  for (int row = tid; row < N; row += 256) {
    size_t gi = (size_t)b * N + row;
    size_t gr = (size_t)b * N + rro[gi];
    float logs = 0.f;
    if (cnt2[gi] > 0 && cnt2[gr] > 0) logs += lse2[gi] + lse2[gr];
    if (cnt4[gi] > 0 && cnt4[gr] > 0) logs += lse4[gi] + lse4[gr];
    if (hpo[gi]) {
      float z = fmaxf(logs + pospair[gi], 0.f);
      accJ += z * z;
      accC += 1;
    }
  }
  sJ[tid] = accJ; sC[tid] = accC;
  __syncthreads();
  for (int s = 128; s > 0; s >>= 1) {
    if (tid < s) { sJ[tid] += sJ[tid + s]; sC[tid] += sC[tid + s]; }
    __syncthreads();
  }
  if (tid == 0) batchloss[b] = (sC[0] > 0) ? sJ[0] / (float)sC[0] : 0.f;
}

// ---------- K4: mean over batches ----------
__global__ void k_final(const float* __restrict__ batchloss, float* __restrict__ out) {
  if (threadIdx.x == 0) {
    float s = 0.f;
    for (int b = 0; b < BATCH; ++b) s += batchloss[b];
    out[0] = s / (float)BATCH;
  }
}

// ---------- host ----------
extern "C" void kernel_launch(void* const* d_in, const int* in_sizes, int n_in,
                              void* d_out, int out_size, void* d_ws, size_t ws_size,
                              hipStream_t stream) {
  (void)in_sizes; (void)n_in; (void)out_size; (void)ws_size;
  const float* x = (const float*)d_in[0];
  const int*   t = (const int*)d_in[1];
  float* out = (float*)d_out;

  char*  ws  = (char*)d_ws;
  size_t off = 0;
  auto alloc = [&](size_t bytes) -> void* {
    void* p = ws + off;
    off += (bytes + 255) & ~(size_t)255;
    return p;
  };
  unsigned short* xhi = (unsigned short*)alloc((size_t)BATCH * N * D * 2);
  unsigned short* xlo = (unsigned short*)alloc((size_t)BATCH * N * D * 2);
  float* sq        = (float*)alloc((size_t)BATCH * N * 4);
  float* lse2      = (float*)alloc((size_t)BATCH * N * 4);
  float* lse4      = (float*)alloc((size_t)BATCH * N * 4);
  int*   cnt2      = (int*)alloc((size_t)BATCH * N * 4);
  int*   cnt4      = (int*)alloc((size_t)BATCH * N * 4);
  float* pospair   = (float*)alloc((size_t)BATCH * N * 4);
  int*   rro       = (int*)alloc((size_t)BATCH * N * 4);
  int*   hpo       = (int*)alloc((size_t)BATCH * N * 4);
  float* batchloss = (float*)alloc((size_t)BATCH * 4);

  const int total = BATCH * N * D;
  k_split<<<(total + 255) / 256, 256, 0, stream>>>(x, xhi, xlo, total);
  k_sqnorm<<<(BATCH * N) / 8, 256, 0, stream>>>(x, sq);
  k_panel<<<BATCH * (N / 16), 256, 16 * N * sizeof(float), stream>>>(
      xhi, xlo, sq, t, lse2, lse4, cnt2, cnt4, pospair, rro, hpo);
  k_combine<<<BATCH, 256, 0, stream>>>(lse2, lse4, cnt2, cnt4, pospair, rro, hpo,
                                       batchloss);
  k_final<<<1, 32, 0, stream>>>(batchloss, out);
}
